// TransformerDecoderLayer_76879914598539
// MI455X (gfx1250) — matrix operand order repaired and checked by reading, not compiled, not run on hardware
//
#include <hip/hip_runtime.h>

typedef __bf16 bf16_t;
typedef __attribute__((ext_vector_type(16))) __bf16 v16bf;
typedef __attribute__((ext_vector_type(8)))  float  v8f;

#define DEVI __device__ __forceinline__

namespace {
constexpr int D_   = 256;
constexpr int H_   = 8;
constexpr int NQ_  = 300;
constexpr int BS_  = 16;
constexpr int HW_  = 1024;
constexpr int DFF_ = 2048;
constexpr int NTOK = NQ_ * BS_;   // 4800
constexpr int BH_  = BS_ * H_;    // 128
constexpr int NQP  = 320;         // padded query count (mult of 32)
constexpr float PI_ = 3.1415926f;
}

DEVI bf16_t f2bf(float x) { return (bf16_t)x; }

// ---------------------------------------------------------------------------
// fp32 -> bf16 conversion
// ---------------------------------------------------------------------------
__global__ void cvt_bf16_kernel(const float* __restrict__ x, bf16_t* __restrict__ y, size_t n) {
    size_t i = (size_t)blockIdx.x * blockDim.x + threadIdx.x;
    if (i < n) y[i] = f2bf(x[i]);
}

// ---------------------------------------------------------------------------
// Rotated-gaussian prior: out[b, q, hw]
// ---------------------------------------------------------------------------
__global__ void gaussian_kernel(const float* __restrict__ rb, const float* __restrict__ grid,
                                const int* __restrict__ mh, const int* __restrict__ mw,
                                float* __restrict__ out) {
    const int q = blockIdx.x;
    const int b = blockIdx.y;
    const float W  = (float)mw[0];
    const float Hh = (float)mh[0];
    const float* r = rb + ((size_t)q * BS_ + b) * 5;
    const float cx = r[0] * W, cy = r[1] * Hh;
    const float w  = r[2] * W, h  = r[3] * Hh;
    const float th = (r[4] - 0.5f) * PI_;
    const float c = cosf(th), s = sinf(th);
    const float ga = w * w * 0.25f, gb = h * h * 0.25f;
    const float m00 = ga * c * c + gb * s * s;
    const float m01 = (gb - ga) * s * c;
    const float m11 = ga * s * s + gb * c * c;
    const float inv = 1.0f / (m00 * m11 - m01 * m01);
    const float si00 = m11 * inv, si01 = -m01 * inv, si11 = m00 * inv;
    float* op = out + ((size_t)b * NQ_ + q) * HW_;
    const float* gp = grid + (size_t)b * HW_ * 2;
    for (int k = threadIdx.x; k < HW_; k += blockDim.x) {
        const float dx = gp[2 * k] - cx;
        const float dy = gp[2 * k + 1] - cy;
        const float quad = si00 * dx * dx + 2.0f * si01 * dx * dy + si11 * dy * dy;
        const float e = __expf(-0.5f * quad);
        op[k] = 1.0f / (1.0f + __expf(-e));
    }
}

// ---------------------------------------------------------------------------
// WMMA fragment loaders (CDNA5 ISA 7.12.2, wave32)
// ---------------------------------------------------------------------------
DEVI v16bf load_a_frag(const bf16_t* __restrict__ A, int lda, int row, int k, int hi) {
    const bf16_t* p = A + (size_t)row * lda + k + hi * 8;
    v16bf r;
    ((uint4*)&r)[0] = *(const uint4*)(p);        // K = k+hi*8    .. +7
    ((uint4*)&r)[1] = *(const uint4*)(p + 16);   // K = k+16+hi*8 .. +7
    return r;
}
// B tile staged in LDS as [64 rows][32 K] bf16 (64B per row)
DEVI v16bf load_b_lds(const bf16_t* sb, int row, int hi) {
    const bf16_t* p = sb + row * 32 + hi * 8;
    v16bf r;
    ((uint4*)&r)[0] = *(const uint4*)(p);        // K = hi*8    .. +7
    ((uint4*)&r)[1] = *(const uint4*)(p + 16);   // K = 16+hi*8 .. +7
    return r;
}

DEVI void async_copy16(unsigned lds_addr, const void* gaddr) {
    // GLOBAL_LOAD_ASYNC_TO_LDS_B128: per-lane 16B global -> LDS, ASYNCcnt-tracked
    asm volatile("global_load_async_to_lds_b128 %0, %1, off"
                 :: "v"(lds_addr), "v"(gaddr)
                 : "memory");
}
DEVI void wait_asynccnt0() {
    asm volatile("s_wait_asynccnt 0" ::: "memory");
}

// ---------------------------------------------------------------------------
// Strided-batched WMMA bf16 GEMM: C[M,N] = act(A[M,K] * B[N,K]^T + bias)
// Block = 4 waves -> 128x64 C tile. Each wave: 32x64 (2 A frags x 4 B frags,
// 8 accumulators). B K-tiles (64x32 = 4KB) double-buffered in LDS via
// async global->LDS copies; drained with s_wait_asynccnt + barrier.
// ---------------------------------------------------------------------------
__global__ void gemm_wmma_bf16(const bf16_t* __restrict__ A, const bf16_t* __restrict__ B,
                               const float* __restrict__ bias,
                               float* __restrict__ Cf, bf16_t* __restrict__ Cb,
                               int M, int N, int K, long sA, long sB, long sC, int relu) {
    __shared__ __align__(16) bf16_t smem[2 * 64 * 32];   // 2 x 4KB B tiles
    const int tid  = threadIdx.x;
    const int wave = tid >> 5;
    const int lane = tid & 31;
    const int hi = lane >> 4;
    const int lo = lane & 15;
    const int m0 = (blockIdx.x * 4 + wave) * 32;
    const int n0 = blockIdx.y * 64;
    const int batch = blockIdx.z;
    A += (size_t)batch * (size_t)sA;
    B += (size_t)batch * (size_t)sB;

    int ar0 = m0 + lo;      if (ar0 >= M) ar0 = M - 1;
    int ar1 = m0 + 16 + lo; if (ar1 >= M) ar1 = M - 1;

    const unsigned lds_base = (unsigned)(uintptr_t)(&smem[0]);

    // stage one 64x32 B tile (256 x 16B chunks; 2 chunks per thread)
    auto stage = [&](int buf, int k) {
        #pragma unroll
        for (int u = 0; u < 2; ++u) {
            const int c = tid * 2 + u;
            const int row = c >> 2, part = c & 3;
            int brow = n0 + row; if (brow >= N) brow = N - 1;
            const bf16_t* g = B + (size_t)brow * K + k + part * 8;
            const unsigned l = lds_base + (unsigned)(buf * 4096 + row * 64 + part * 16);
            async_copy16(l, g);
        }
    };

    const v8f vzero = {0, 0, 0, 0, 0, 0, 0, 0};
    v8f acc[2][4];
    #pragma unroll
    for (int i = 0; i < 2; ++i)
        #pragma unroll
        for (int j = 0; j < 4; ++j) acc[i][j] = vzero;

    stage(0, 0);
    wait_asynccnt0();
    __syncthreads();

    int buf = 0;
    for (int k = 0; k < K; k += 32) {
        if (k + 32 < K) stage(buf ^ 1, k + 32);
        v16bf a0 = load_a_frag(A, K, ar0, k, hi);
        v16bf a1 = load_a_frag(A, K, ar1, k, hi);
        const bf16_t* sb = &smem[buf * 2048];
        #pragma unroll
        for (int j = 0; j < 4; ++j) {
            v16bf bj = load_b_lds(sb, j * 16 + lo, hi);
            acc[0][j] = __builtin_amdgcn_wmma_f32_16x16x32_bf16(false, a0, false, bj, (short)0, acc[0][j], false, false);
            acc[1][j] = __builtin_amdgcn_wmma_f32_16x16x32_bf16(false, a1, false, bj, (short)0, acc[1][j], false, false);
        }
        wait_asynccnt0();
        __syncthreads();
        buf ^= 1;
    }

    #pragma unroll
    for (int i = 0; i < 2; ++i) {
        const int rbase = m0 + i * 16 + hi * 8;
        #pragma unroll
        for (int j = 0; j < 4; ++j) {
            const int cn = n0 + j * 16 + lo;
            if (cn >= N) continue;
            const float bv = bias ? bias[cn] : 0.0f;
            for (int r = 0; r < 8; ++r) {
                const int row = rbase + r;
                if (row >= M) break;
                float v = acc[i][j][r] + bv;
                if (relu && v < 0.0f) v = 0.0f;
                const size_t idx = (size_t)batch * (size_t)sC + (size_t)row * N + cn;
                if (Cf) Cf[idx] = v;
                if (Cb) Cb[idx] = f2bf(v);
            }
        }
    }
}

// ---------------------------------------------------------------------------
// Pack self-attention Q (scaled), K, V^T per (b,h); pad queries 300 -> 320
// ---------------------------------------------------------------------------
__global__ void pack_self_qkv(const float* __restrict__ tgt, const float* __restrict__ qp,
                              bf16_t* __restrict__ Qb, bf16_t* __restrict__ Kb,
                              bf16_t* __restrict__ Vt) {
    const size_t n = (size_t)BH_ * NQP * 32;
    size_t i = (size_t)blockIdx.x * blockDim.x + threadIdx.x;
    if (i >= n) return;
    const int d  = (int)(i & 31);
    const int q  = (int)((i >> 5) % NQP);
    const int bh = (int)(i / ((size_t)NQP * 32));
    const int b = bh >> 3, h = bh & 7;
    float qv = 0.f, kv = 0.f, vv = 0.f;
    if (q < NQ_) {
        const size_t ridx = ((size_t)q * BS_ + b) * D_ + h * 32 + d;
        const float t = tgt[ridx];
        kv = t + qp[ridx];
        qv = kv * 0.17677669529663687f;   // 32^-0.5
        vv = t;
    }
    Qb[i] = f2bf(qv);
    Kb[i] = f2bf(kv);
    Vt[((size_t)bh * 32 + d) * NQP + q] = f2bf(vv);
}

// ---------------------------------------------------------------------------
// In-place row softmax over bf16 logits; optional multiplicative gaussian
// S: [BH, NQ, Lpad]; gauss: [BS, NQ, Lpad] (Lpad == HW for cross-attn)
// ---------------------------------------------------------------------------
__global__ void softmax_rows(bf16_t* __restrict__ S, const float* __restrict__ gauss,
                             int Lvalid, int Lpad) {
    const int row = blockIdx.x;            // bh*NQ + q
    const int bh = row / NQ_;
    const int q  = row - bh * NQ_;
    bf16_t* sp = S + (size_t)row * Lpad;
    const float* gp = gauss ? (gauss + ((size_t)(bh >> 3) * NQ_ + q) * (size_t)Lpad) : nullptr;
    const int t = threadIdx.x;
    const int nt = blockDim.x;
    const int nw = nt >> 5;
    __shared__ float sred[8];

    float lmax = -3.0e38f;
    for (int c = t; c < Lvalid; c += nt) {
        float x = (float)sp[c];
        if (gp) x *= gp[c];
        lmax = fmaxf(lmax, x);
    }
    for (int o = 16; o > 0; o >>= 1) lmax = fmaxf(lmax, __shfl_xor(lmax, o, 32));
    if ((t & 31) == 0) sred[t >> 5] = lmax;
    __syncthreads();
    float m = -3.0e38f;
    for (int i = 0; i < nw; ++i) m = fmaxf(m, sred[i]);
    __syncthreads();

    float lsum = 0.f;
    for (int c = t; c < Lvalid; c += nt) {
        float x = (float)sp[c];
        if (gp) x *= gp[c];
        lsum += __expf(x - m);
    }
    for (int o = 16; o > 0; o >>= 1) lsum += __shfl_xor(lsum, o, 32);
    if ((t & 31) == 0) sred[t >> 5] = lsum;
    __syncthreads();
    float sum = 0.f;
    for (int i = 0; i < nw; ++i) sum += sred[i];
    const float rinv = 1.0f / sum;

    for (int c = t; c < Lpad; c += nt) {
        float v = 0.0f;
        if (c < Lvalid) {
            float x = (float)sp[c];
            if (gp) x *= gp[c];
            v = __expf(x - m) * rinv;
        }
        sp[c] = f2bf(v);
    }
}

// ---------------------------------------------------------------------------
// [bh][q][d] fp32 head context -> token-major bf16 [ (q*BS+b)*D + h*32+d ]
// ---------------------------------------------------------------------------
__global__ void unpack_heads(const float* __restrict__ headC, bf16_t* __restrict__ ctx_bf) {
    const size_t n = (size_t)BH_ * NQ_ * 32;
    size_t i = (size_t)blockIdx.x * blockDim.x + threadIdx.x;
    if (i >= n) return;
    const int d  = (int)(i & 31);
    const int q  = (int)((i >> 5) % NQ_);
    const int bh = (int)(i / ((size_t)NQ_ * 32));
    const int b = bh >> 3, h = bh & 7;
    ctx_bf[((size_t)q * BS_ + b) * D_ + h * 32 + d] = f2bf(headC[i]);
}

// ---------------------------------------------------------------------------
// LayerNorm(x + res) * g + b ; one block (256 threads) per row
// ---------------------------------------------------------------------------
__global__ void ln_kernel(const float* __restrict__ x, const float* __restrict__ res,
                          const float* __restrict__ g, const float* __restrict__ bta,
                          float* __restrict__ yf, bf16_t* __restrict__ yb) {
    const int row = blockIdx.x;
    const int t = threadIdx.x;
    __shared__ float sred[8];
    float v = x[(size_t)row * D_ + t];
    if (res) v += res[(size_t)row * D_ + t];
    float s = v;
    for (int o = 16; o > 0; o >>= 1) s += __shfl_xor(s, o, 32);
    if ((t & 31) == 0) sred[t >> 5] = s;
    __syncthreads();
    float mean = 0.f;
    for (int i = 0; i < 8; ++i) mean += sred[i];
    mean *= (1.0f / D_);
    __syncthreads();
    const float dv = v - mean;
    float s2 = dv * dv;
    for (int o = 16; o > 0; o >>= 1) s2 += __shfl_xor(s2, o, 32);
    if ((t & 31) == 0) sred[t >> 5] = s2;
    __syncthreads();
    float var = 0.f;
    for (int i = 0; i < 8; ++i) var += sred[i];
    var *= (1.0f / D_);
    const float y = dv * rsqrtf(var + 1e-5f) * g[t] + bta[t];
    if (yf) yf[(size_t)row * D_ + t] = y;
    if (yb) yb[(size_t)row * D_ + t] = f2bf(y);
}

// ---------------------------------------------------------------------------
// Cross-attention packing: Qc [bh][300][64] (scaled), Kc [bh][1024][64],
// Vmt [bh][32][1024]
// ---------------------------------------------------------------------------
__global__ void pack_cross_q(const float* __restrict__ tgt1, const float* __restrict__ qrbse,
                             const float* __restrict__ qcm, bf16_t* __restrict__ Qc) {
    const size_t n = (size_t)BH_ * NQ_ * 64;
    size_t i = (size_t)blockIdx.x * blockDim.x + threadIdx.x;
    if (i >= n) return;
    const int e  = (int)(i & 63);
    const int q  = (int)((i >> 6) % NQ_);
    const int bh = (int)(i / ((size_t)NQ_ * 64));
    const int b = bh >> 3, h = bh & 7;
    const size_t row = (size_t)q * BS_ + b;
    float v;
    if (e < 32) {
        v = tgt1[row * D_ + h * 32 + e];
    } else if (e < 48) {
        const int c = h * 16 + (e - 32);
        v = qrbse[row * 640 + c] * qcm[row * D_ + c];
    } else {
        const int c = h * 16 + (e - 48);
        v = qrbse[row * 640 + 128 + c] * qcm[row * D_ + 128 + c];
    }
    Qc[i] = f2bf(v * 0.125f);   // 64^-0.5
}

__global__ void pack_cross_k(const float* __restrict__ memory, const float* __restrict__ pos,
                             bf16_t* __restrict__ Kc) {
    const size_t n = (size_t)BH_ * HW_ * 64;
    size_t i = (size_t)blockIdx.x * blockDim.x + threadIdx.x;
    if (i >= n) return;
    const int e  = (int)(i & 63);
    const int k  = (int)((i >> 6) % HW_);
    const int bh = (int)(i / ((size_t)HW_ * 64));
    const int b = bh >> 3, h = bh & 7;
    const size_t row = (size_t)k * BS_ + b;
    float v;
    if (e < 32)       v = memory[row * D_ + h * 32 + e];
    else if (e < 48)  v = pos[row * D_ + h * 16 + (e - 32)];
    else              v = pos[row * D_ + 128 + h * 16 + (e - 48)];
    Kc[i] = f2bf(v);
}

__global__ void pack_cross_v(const float* __restrict__ memory, bf16_t* __restrict__ Vmt) {
    const size_t n = (size_t)BH_ * 32 * HW_;
    size_t i = (size_t)blockIdx.x * blockDim.x + threadIdx.x;
    if (i >= n) return;
    const int k  = (int)(i & (HW_ - 1));
    const int d  = (int)((i >> 10) & 31);
    const int bh = (int)(i / ((size_t)32 * HW_));
    const int b = bh >> 3, h = bh & 7;
    Vmt[i] = f2bf(memory[((size_t)k * BS_ + b) * D_ + h * 32 + d]);
}

// ---------------------------------------------------------------------------
// Host orchestration
// ---------------------------------------------------------------------------
extern "C" void kernel_launch(void* const* d_in, const int* in_sizes, int n_in,
                              void* d_out, int out_size, void* d_ws, size_t ws_size,
                              hipStream_t stream) {
    (void)in_sizes; (void)n_in; (void)out_size; (void)ws_size;
    const float* tgt      = (const float*)d_in[0];
    const float* memory   = (const float*)d_in[1];
    const float* pos      = (const float*)d_in[2];
    const float* qrbse    = (const float*)d_in[3];
    const float* refb     = (const float*)d_in[4];
    const float* grid     = (const float*)d_in[5];
    const float* sa_w1    = (const float*)d_in[6];
    const float* sa_b1    = (const float*)d_in[7];
    const float* sa_w2    = (const float*)d_in[8];
    const float* sa_b2    = (const float*)d_in[9];
    const float* sa_out_w = (const float*)d_in[10];
    const float* sa_out_b = (const float*)d_in[11];
    const float* ln1_g    = (const float*)d_in[12];
    const float* ln1_b    = (const float*)d_in[13];
    const float* cam_w1   = (const float*)d_in[14];
    const float* cam_b1   = (const float*)d_in[15];
    const float* cam_w2   = (const float*)d_in[16];
    const float* cam_b2   = (const float*)d_in[17];
    const float* ca_out_w = (const float*)d_in[18];
    const float* ca_out_b = (const float*)d_in[19];
    const float* ln2_g    = (const float*)d_in[20];
    const float* ln2_b    = (const float*)d_in[21];
    const float* lin1_w   = (const float*)d_in[22];
    const float* lin1_b   = (const float*)d_in[23];
    const float* lin2_w   = (const float*)d_in[24];
    const float* lin2_b   = (const float*)d_in[25];
    const float* ln3_g    = (const float*)d_in[26];
    const float* ln3_b    = (const float*)d_in[27];
    const float* off_w1   = (const float*)d_in[28];
    const float* off_b1   = (const float*)d_in[29];
    const float* off_w2   = (const float*)d_in[30];
    const float* off_b2   = (const float*)d_in[31];
    const int*   mh       = (const int*)d_in[32];
    const int*   mw       = (const int*)d_in[33];
    float* out = (float*)d_out;

    // -------- workspace bump allocator --------
    char* p = (char*)d_ws;
    auto alloc = [&](size_t bytes) -> char* {
        char* r = p;
        p += (bytes + 255) & ~(size_t)255;
        return r;
    };
    float*  g_gauss  = (float*)alloc((size_t)BS_ * NQ_ * HW_ * 4);
    bf16_t* w_sa1b   = (bf16_t*)alloc((size_t)D_ * 640 * 2);
    bf16_t* w_sa2b   = (bf16_t*)alloc((size_t)D_ * D_ * 2);
    bf16_t* w_saoutb = (bf16_t*)alloc((size_t)D_ * D_ * 2);
    bf16_t* w_cam1b  = (bf16_t*)alloc((size_t)D_ * D_ * 2);
    bf16_t* w_cam2b  = (bf16_t*)alloc((size_t)D_ * D_ * 2);
    bf16_t* w_caoutb = (bf16_t*)alloc((size_t)D_ * D_ * 2);
    bf16_t* w_lin1b  = (bf16_t*)alloc((size_t)DFF_ * D_ * 2);
    bf16_t* w_lin2b  = (bf16_t*)alloc((size_t)D_ * DFF_ * 2);
    bf16_t* w_off1b  = (bf16_t*)alloc((size_t)D_ * D_ * 2);
    bf16_t* w_off2b  = (bf16_t*)alloc((size_t)5 * D_ * 2);
    bf16_t* qrbse_bf = (bf16_t*)alloc((size_t)NTOK * 640 * 2);
    bf16_t* h1_bf    = (bf16_t*)alloc((size_t)NTOK * D_ * 2);
    float*  qp_f     = (float*)alloc((size_t)NTOK * D_ * 4);
    bf16_t* Qb       = (bf16_t*)alloc((size_t)BH_ * NQP * 32 * 2);
    bf16_t* Kb       = (bf16_t*)alloc((size_t)BH_ * NQP * 32 * 2);
    bf16_t* Vt       = (bf16_t*)alloc((size_t)BH_ * 32 * NQP * 2);
    bf16_t* S_self   = (bf16_t*)alloc((size_t)BH_ * NQ_ * NQP * 2);
    float*  sa_head  = (float*)alloc((size_t)BH_ * NQ_ * 32 * 4);
    bf16_t* sa_ctxb  = (bf16_t*)alloc((size_t)NTOK * D_ * 2);
    float*  sa_proj  = (float*)alloc((size_t)NTOK * D_ * 4);
    float*  tgt1_f   = (float*)alloc((size_t)NTOK * D_ * 4);
    bf16_t* tgt1_bf  = (bf16_t*)alloc((size_t)NTOK * D_ * 2);
    bf16_t* h2_bf    = (bf16_t*)alloc((size_t)NTOK * D_ * 2);
    float*  qcm_f    = (float*)alloc((size_t)NTOK * D_ * 4);
    bf16_t* Qc       = (bf16_t*)alloc((size_t)BH_ * NQ_ * 64 * 2);
    bf16_t* Kc       = (bf16_t*)alloc((size_t)BH_ * HW_ * 64 * 2);
    bf16_t* Vmt      = (bf16_t*)alloc((size_t)BH_ * 32 * HW_ * 2);
    bf16_t* S_cross  = (bf16_t*)alloc((size_t)BH_ * NQ_ * HW_ * 2);
    float*  ca_head  = (float*)alloc((size_t)BH_ * NQ_ * 32 * 4);
    bf16_t* ca_ctxb  = (bf16_t*)alloc((size_t)NTOK * D_ * 2);
    float*  ca_proj  = (float*)alloc((size_t)NTOK * D_ * 4);
    float*  tgt2_f   = (float*)alloc((size_t)NTOK * D_ * 4);
    bf16_t* tgt2_bf  = (bf16_t*)alloc((size_t)NTOK * D_ * 2);
    bf16_t* ffn1_bf  = (bf16_t*)alloc((size_t)NTOK * DFF_ * 2);
    float*  ffn2_f   = (float*)alloc((size_t)NTOK * D_ * 4);
    bf16_t* tgt3_bf  = (bf16_t*)alloc((size_t)NTOK * D_ * 2);
    bf16_t* offh_bf  = (bf16_t*)alloc((size_t)NTOK * D_ * 2);

    auto cvt = [&](const float* x, bf16_t* y, size_t n) {
        cvt_bf16_kernel<<<dim3((unsigned)((n + 255) / 256)), dim3(256), 0, stream>>>(x, y, n);
    };
    auto gemm = [&](const bf16_t* A, const bf16_t* B, const float* bias,
                    float* Cf, bf16_t* Cb, int M, int N, int K,
                    long sA, long sB, long sC, int relu, int batch) {
        dim3 g((unsigned)((M + 127) / 128), (unsigned)((N + 63) / 64), (unsigned)batch);
        gemm_wmma_bf16<<<g, dim3(128), 0, stream>>>(A, B, bias, Cf, Cb, M, N, K, sA, sB, sC, relu);
    };
    auto elems = [&](size_t n) { return dim3((unsigned)((n + 255) / 256)); };

    // -------- weight / input conversions --------
    cvt(sa_w1, w_sa1b, (size_t)D_ * 640);
    cvt(sa_w2, w_sa2b, (size_t)D_ * D_);
    cvt(sa_out_w, w_saoutb, (size_t)D_ * D_);
    cvt(cam_w1, w_cam1b, (size_t)D_ * D_);
    cvt(cam_w2, w_cam2b, (size_t)D_ * D_);
    cvt(ca_out_w, w_caoutb, (size_t)D_ * D_);
    cvt(lin1_w, w_lin1b, (size_t)DFF_ * D_);
    cvt(lin2_w, w_lin2b, (size_t)D_ * DFF_);
    cvt(off_w1, w_off1b, (size_t)D_ * D_);
    cvt(off_w2, w_off2b, (size_t)5 * D_);
    cvt(qrbse, qrbse_bf, (size_t)NTOK * 640);

    // -------- gaussian prior --------
    gaussian_kernel<<<dim3(NQ_, BS_), dim3(256), 0, stream>>>(refb, grid, mh, mw, g_gauss);

    // -------- query_pos MLP --------
    gemm(qrbse_bf, w_sa1b, sa_b1, nullptr, h1_bf, NTOK, D_, 640, 0, 0, 0, 1, 1);
    gemm(h1_bf, w_sa2b, sa_b2, qp_f, nullptr, NTOK, D_, D_, 0, 0, 0, 0, 1);

    // -------- self-attention --------
    {
        size_t n = (size_t)BH_ * NQP * 32;
        pack_self_qkv<<<elems(n), dim3(256), 0, stream>>>(tgt, qp_f, Qb, Kb, Vt);
    }
    gemm(Qb, Kb, nullptr, nullptr, S_self, NQ_, NQP, 32,
         (long)NQP * 32, (long)NQP * 32, (long)NQ_ * NQP, 0, BH_);
    softmax_rows<<<dim3(BH_ * NQ_), dim3(128), 0, stream>>>(S_self, nullptr, NQ_, NQP);
    gemm(S_self, Vt, nullptr, sa_head, nullptr, NQ_, 32, NQP,
         (long)NQ_ * NQP, (long)32 * NQP, (long)NQ_ * 32, 0, BH_);
    {
        size_t n = (size_t)BH_ * NQ_ * 32;
        unpack_heads<<<elems(n), dim3(256), 0, stream>>>(sa_head, sa_ctxb);
    }
    gemm(sa_ctxb, w_saoutb, sa_out_b, sa_proj, nullptr, NTOK, D_, D_, 0, 0, 0, 0, 1);
    ln_kernel<<<dim3(NTOK), dim3(D_), 0, stream>>>(sa_proj, tgt, ln1_g, ln1_b, tgt1_f, tgt1_bf);

    // -------- cam MLP --------
    gemm(tgt1_bf, w_cam1b, cam_b1, nullptr, h2_bf, NTOK, D_, D_, 0, 0, 0, 1, 1);
    gemm(h2_bf, w_cam2b, cam_b2, qcm_f, nullptr, NTOK, D_, D_, 0, 0, 0, 0, 1);

    // -------- cross-attention --------
    {
        size_t nq = (size_t)BH_ * NQ_ * 64;
        pack_cross_q<<<elems(nq), dim3(256), 0, stream>>>(tgt1_f, qrbse, qcm_f, Qc);
        size_t nk = (size_t)BH_ * HW_ * 64;
        pack_cross_k<<<elems(nk), dim3(256), 0, stream>>>(memory, pos, Kc);
        size_t nv = (size_t)BH_ * 32 * HW_;
        pack_cross_v<<<elems(nv), dim3(256), 0, stream>>>(memory, Vmt);
    }
    gemm(Qc, Kc, nullptr, nullptr, S_cross, NQ_, HW_, 64,
         (long)NQ_ * 64, (long)HW_ * 64, (long)NQ_ * HW_, 0, BH_);
    softmax_rows<<<dim3(BH_ * NQ_), dim3(256), 0, stream>>>(S_cross, g_gauss, HW_, HW_);
    gemm(S_cross, Vmt, nullptr, ca_head, nullptr, NQ_, 32, HW_,
         (long)NQ_ * HW_, (long)32 * HW_, (long)NQ_ * 32, 0, BH_);
    {
        size_t n = (size_t)BH_ * NQ_ * 32;
        unpack_heads<<<elems(n), dim3(256), 0, stream>>>(ca_head, ca_ctxb);
    }
    gemm(ca_ctxb, w_caoutb, ca_out_b, ca_proj, nullptr, NTOK, D_, D_, 0, 0, 0, 0, 1);
    ln_kernel<<<dim3(NTOK), dim3(D_), 0, stream>>>(ca_proj, tgt1_f, ln2_g, ln2_b, tgt2_f, tgt2_bf);

    // -------- FFN --------
    gemm(tgt2_bf, w_lin1b, lin1_b, nullptr, ffn1_bf, NTOK, DFF_, D_, 0, 0, 0, 1, 1);
    gemm(ffn1_bf, w_lin2b, lin2_b, ffn2_f, nullptr, NTOK, D_, DFF_, 0, 0, 0, 0, 1);
    ln_kernel<<<dim3(NTOK), dim3(D_), 0, stream>>>(ffn2_f, tgt2_f, ln3_g, ln3_b, out, tgt3_bf);

    // -------- offset head --------
    gemm(tgt3_bf, w_off1b, off_b1, nullptr, offh_bf, NTOK, D_, D_, 0, 0, 0, 1, 1);
    gemm(offh_bf, w_off2b, off_b2, out + (size_t)NTOK * D_, nullptr,
         NTOK, 5, D_, 0, 0, 0, 0, 1);
}
